// DTModel_36696200577414
// MI455X (gfx1250) — matrix operand order repaired and checked by reading, not compiled, not run on hardware
//
#include <hip/hip_runtime.h>

// Soft decision tree, DEPTH=8, 32 feats, 10 classes, B=131072.
// Phase 1: fully register-resident tree expansion; sigmoid via single-trans
//          v_tanh_f32 (sigma(z) = 0.5*tanh(z/2)+0.5), x staged half-scaled in LDS.
// Phase 2: leaf probs packed to f16 in LDS, per-wave WMMA reduction
//          probs(16x256) @ onehot(256x16) via v_wmma_f32_16x16x32_f16.

#define DEPTH    8
#define NFEAT    32
#define NCLS     10
#define NLEAF    256
#define ROWS     128
#define THREADS  128
#define PH_STRIDE (NLEAF + 2)   // 258 halves = 129 dwords/row -> bank (r+i)%64

typedef __attribute__((ext_vector_type(16))) _Float16 v16h;
typedef __attribute__((ext_vector_type(2)))  _Float16 h2;
typedef __attribute__((ext_vector_type(8)))  float    v8f;

// sigmoid(2d) as a function of d = (x - t)/2.
// Device side: UNCONDITIONAL v_tanh_f32 probe — if this builtin name is wrong
// for gfx1250, the compile fails loudly and we revert to the exp+rcp form.
__device__ __forceinline__ float sigmoid_from_half(float d) {
#if defined(__AMDGCN__)
    return __builtin_fmaf(0.5f, __builtin_amdgcn_tanhf(d), 0.5f);   // 1 trans op
#else
    return 0.5f * (1.0f + d);  // host pass placeholder, never executed
#endif
}

__global__ __launch_bounds__(THREADS) void dtree_kernel(
    const float* __restrict__ x,       // [B][32]
    const float* __restrict__ thr,     // [255]
    const int*   __restrict__ feats,   // [255]
    const int*   __restrict__ leafc,   // [256]
    float*       __restrict__ out)     // [B][10]
{
    // xs stride 33 floats -> bank (33r+f)%64 distinct across lanes.
    __shared__ float    xs[ROWS][NFEAT + 1];   // holds 0.5*x
    __shared__ _Float16 ph[ROWS][PH_STRIDE];   // leaf probs, f16

    const int  r    = threadIdx.x;
    const long gRow = (long)blockIdx.x * ROWS + r;

    // ---- load this thread's x row into LDS, pre-scaled by 0.5 ----
    {
        const float4* src = (const float4*)(x + gRow * NFEAT);
        #pragma unroll
        for (int i = 0; i < NFEAT / 4; ++i) {
            float4 v = src[i];
            xs[r][4 * i + 0] = 0.5f * v.x;
            xs[r][4 * i + 1] = 0.5f * v.y;
            xs[r][4 * i + 2] = 0.5f * v.z;
            xs[r][4 * i + 3] = 0.5f * v.w;
        }
    }

    // ---- tree expansion, fully unrolled, probs live in VGPRs (SROA) ----
    // per node: v_fma (d), v_tanh, v_fma (a), v_mul (pr), v_sub (pl)
    float p[NLEAF];
    p[0] = 1.0f;
    int start = 0;
    #pragma unroll
    for (int d = 0; d < DEPTH; ++d) {
        const int n = 1 << d;
        #pragma unroll
        for (int j = n - 1; j >= 0; --j) {           // descending j: in-place safe
            const int   f = feats[start + j];        // uniform -> s_load
            const float t = thr[start + j];          // uniform -> s_load
            const float dh = __builtin_fmaf(t, -0.5f, xs[r][f]);  // (x - t)/2
            const float a  = sigmoid_from_half(dh);
            const float pp = p[j];
            const float pr = pp * a;
            p[2 * j + 1] = pr;
            p[2 * j]     = pp - pr;                  // pp*(1-a) without extra op
        }
        start += n;
    }

    // ---- pack leaves to f16 pairs, 128 conflict-free dword stores ----
    #pragma unroll
    for (int i = 0; i < NLEAF / 2; ++i) {
        h2 v;
        v[0] = (_Float16)p[2 * i];
        v[1] = (_Float16)p[2 * i + 1];
        *(h2*)&ph[r][2 * i] = v;                     // v_cvt_pk_f16_f32 + ds_store_b32
    }

    __syncthreads();

    // ---- WMMA reduction: probs @ onehot(leaf_class), f16 A/B, f32 C ----
    const int lane   = threadIdx.x & 31;
    const int wave   = threadIdx.x >> 5;
    const int laneN  = lane & 15;
    const int laneHi = lane >> 4;

    const _Float16 h1 = (_Float16)1.0f;
    const _Float16 hz = (_Float16)0.0f;

    // B layout (contiguous k order used here, same as A): lane n(0-15): N=n, K=chunk*32 + 0..15;
    // lanes 16-31: K=chunk*32 + 16..31.  Element kk <-> K = laneHi*16 + kk.
    v16h bfrag[8];
    #pragma unroll
    for (int c = 0; c < 8; ++c) {
        v16h b;
        #pragma unroll
        for (int kk = 0; kk < 16; ++kk) {
            const int c0  = leafc[c * 32 + kk];       // uniform -> s_load
            const int c1  = leafc[c * 32 + 16 + kk];  // uniform -> s_load
            const int cls = laneHi ? c1 : c0;
            b[kk] = (cls == laneN) ? h1 : hz;
        }
        bfrag[c] = b;
    }

    // 8 row-tiles of 16, 4 waves -> 2 tiles per wave.
    float accall[2][8];
    #pragma unroll
    for (int tt = 0; tt < 2; ++tt) {
        const int tile = wave * 2 + tt;
        const int m    = tile * 16 + laneN;           // LDS row feeding A
        v8f acc = {};
        #pragma unroll
        for (int c = 0; c < 8; ++c) {
            // A layout: lane m half-rows: K = laneHi*16 + {0..15}
            const int k0 = c * 32 + laneHi * 16;
            union { v16h h; unsigned int u[8]; } af;
            #pragma unroll
            for (int i = 0; i < 4; ++i) {
                af.u[i]     = *(const unsigned int*)&ph[m][k0 + 2 * i];
                af.u[4 + i] = *(const unsigned int*)&ph[m][k0 + 8 + 2 * i];
            }
            acc = __builtin_amdgcn_wmma_f32_16x16x32_f16(
                false, af.h, false, bfrag[c], (short)0, acc, false, false);
        }
        #pragma unroll
        for (int v = 0; v < 8; ++v) accall[tt][v] = acc[v];
    }

    // Divergent stores strictly after all WMMAs (EXEC all-ones during WMMA).
    if (laneN < NCLS) {
        #pragma unroll
        for (int tt = 0; tt < 2; ++tt) {
            const int  tile    = wave * 2 + tt;
            const long rowBase = (long)blockIdx.x * ROWS + tile * 16;
            #pragma unroll
            for (int v = 0; v < 8; ++v) {
                const long row = rowBase + v + laneHi * 8;   // C: M = v + 8*laneHi
                out[row * NCLS + laneN] = accall[tt][v];
            }
        }
    }
}

extern "C" void kernel_launch(void* const* d_in, const int* in_sizes, int n_in,
                              void* d_out, int out_size, void* d_ws, size_t ws_size,
                              hipStream_t stream) {
    const float* x     = (const float*)d_in[0];
    const float* thr   = (const float*)d_in[1];
    const int*   feats = (const int*)d_in[2];
    const int*   leafc = (const int*)d_in[3];
    float*       out   = (float*)d_out;

    const int batch = in_sizes[0] / NFEAT;   // 131072
    dim3 grid(batch / ROWS);                 // 1024 blocks of 128 threads (4 waves)
    dtree_kernel<<<grid, THREADS, 0, stream>>>(x, thr, feats, leafc, out);
}
